// Converter_80839874445812
// MI455X (gfx1250) — compile-verified, matrix operand order checked by reading, not run on hardware
//
#include <hip/hip_runtime.h>
#include <hip/hip_bf16.h>

typedef __attribute__((ext_vector_type(16))) _Float16 v16h;
typedef __attribute__((ext_vector_type(8)))  float    v8f;

#define B_   256
#define D_   256
#define H_   8
#define L_   4
#define T_   64
#define EPS_ 0.001f

__device__ __forceinline__ float sigmoidf_(float x){ return 1.0f/(1.0f+__expf(-x)); }
__device__ __forceinline__ float tanhf_(float x){ float e=__expf(2.0f*x); return (e-1.0f)/(e+1.0f); }

// Load a 16x32 f16 fragment (A-layout per CDNA5 ISA 7.12.2) from LDS.
// lane<16: M=lane, K-groups {0..7,16..23}; lane>=16: M=lane-16, K-groups {8..15,24..31}
__device__ __forceinline__ void load_frag_a(const _Float16* lds, int pitch, int row0, int kbase, v16h& f){
  int lane = threadIdx.x & 31;
  int m = row0 + (lane & 15);
  int kb = (lane < 16) ? 0 : 8;
  const _Float16* p = lds + m*pitch;
  #pragma unroll
  for (int v = 0; v < 8; ++v){
    int kk = kbase + ((v>>2)<<4) + kb + ((v&3)<<1);
    f[2*v]   = p[kk];
    f[2*v+1] = p[kk+1];
  }
}

// ---------------------------------------------------------------------------
// Generic WMMA GEMM: C[b,ri,n] = sum_c A(b,ri,c) * W[c,n] + bias[n]
// mode 0: A addr = a_off + b*a_sB + ri*a_sR + c*a_sC   (supports transpose/reverse)
// mode 2: conv im2col gather from (b, 64, 256) activation tensor
// Tile 128x64, 8 waves, K-step 32. Register-staged double-buffered LDS,
// per-thread row decomposition hoisted out of the K loop, conv kp->t LUT in LDS.
// ---------------------------------------------------------------------------
struct GemmP {
  const float* A; const float* W; const float* bias; float* C;
  int M, N, Kd, rpb;
  int mode;
  int a_off, a_sB, a_sR, a_sC;
  int ks, os;
  int c_off, c_sB, c_sR;
};

__global__ __launch_bounds__(256) void gemm_wmma(GemmP p){
  __shared__ _Float16 AsBuf[2*128*32];
  __shared__ _Float16 BsBuf[2*64*32];
  __shared__ int tlut[64];

  int tid  = threadIdx.x;
  int wave = tid >> 5;
  int row0 = blockIdx.y * 128;
  int n0   = blockIdx.x * 64;

  // ---- hoisted per-thread addressing (each thread owns 1 A row x 16 cols) ----
  int lr = tid >> 1;                 // A tile row 0..127
  int colBase = (tid & 1) * 16;      // A tile col base 0/16
  int rA = row0 + lr;
  int bA = rA / p.rpb; int riA = rA - bA*p.rpb;
  const float* aBase;
  int tbase = 0;
  if (p.mode == 0){
    aBase = p.A + p.a_off + bA*p.a_sB + riA*p.a_sR;
  } else {
    int os = p.os;
    int px = riA % os; int t1 = riA / os; int py = t1 % os; int pz = t1 / os;
    tbase = pz*16 + py*4 + px;
    aBase = p.A + bA*(T_*D_);
    int kv = p.ks*p.ks*p.ks;
    if (tid < kv){
      int kx = tid % p.ks; int t2 = tid / p.ks; int ky = t2 % p.ks; int kz = t2 / p.ks;
      tlut[tid] = kz*16 + ky*4 + kx;
    }
  }
  int nB = tid >> 2;                 // B tile row (= output col) 0..63
  int kB = (tid & 3) * 8;            // B tile k base
  const float* wBase = p.W + (n0 + nB);

  if (p.mode == 2) __syncthreads();  // tlut ready

  v8f acc[4];
  v8f zero = {0.f,0.f,0.f,0.f,0.f,0.f,0.f,0.f};
  #pragma unroll
  for (int i=0;i<4;++i) acc[i]=zero;

  float aReg[16]; float bReg[8];

  auto loadTile = [&](int kk){
    if (p.mode == 0){
      const float* src = aBase + (size_t)(kk + colBase)*p.a_sC;
      #pragma unroll
      for (int e=0;e<16;++e) aReg[e] = src[(size_t)e*p.a_sC];
    } else {
      int c0 = kk + colBase;
      const float* src = aBase + (tbase + tlut[c0 >> 8])*256 + (c0 & 255);
      #pragma unroll
      for (int e=0;e<16;++e) aReg[e] = src[e];
    }
    const float* wsrc = wBase + (size_t)(kk + kB)*p.N;
    #pragma unroll
    for (int e=0;e<8;++e) bReg[e] = wsrc[(size_t)e*p.N];
  };
  auto storeTile = [&](int buf){
    _Float16* As = AsBuf + buf*(128*32);
    _Float16* Bs = BsBuf + buf*(64*32);
    #pragma unroll
    for (int e=0;e<16;++e) As[lr*32 + colBase + e] = (_Float16)aReg[e];
    #pragma unroll
    for (int e=0;e<8;++e)  Bs[nB*32 + kB + e]      = (_Float16)bReg[e];
  };

  int nk = p.Kd >> 5;
  loadTile(0); storeTile(0);
  __syncthreads();

  for (int kt = 0; kt < nk; ++kt){
    int cur = kt & 1;
    if (kt + 1 < nk) loadTile((kt+1) << 5);
    if (kt + 2 < nk){
      // prefetch A stream two tiles ahead (global_prefetch_b8)
      if (p.mode == 0) __builtin_prefetch(aBase + (size_t)(((kt+2)<<5) + colBase)*p.a_sC, 0, 1);
    }
    const _Float16* As = AsBuf + cur*(128*32);
    const _Float16* Bs = BsBuf + cur*(64*32);
    v16h a; load_frag_a(As, 32, wave*16, 0, a);
    #pragma unroll
    for (int nt = 0; nt < 4; ++nt){
      v16h bf; load_frag_a(Bs, 32, nt*16, 0, bf);
      acc[nt] = __builtin_amdgcn_wmma_f32_16x16x32_f16(false, a, false, bf, (short)0, acc[nt], false, false);
    }
    if (kt + 1 < nk) storeTile(cur ^ 1);
    __syncthreads();
  }

  int lane = tid & 31;
  int colb = lane & 15;
  int rbase = wave*16 + ((lane < 16) ? 0 : 8);
  #pragma unroll
  for (int r = 0; r < 8; ++r){
    int row = row0 + rbase + r;
    int b = row / p.rpb; int ri = row - b*p.rpb;
    float* crow = p.C + p.c_off + b*p.c_sB + ri*p.c_sR;
    #pragma unroll
    for (int nt = 0; nt < 4; ++nt){
      int n = n0 + nt*16 + colb;
      float bv = p.bias ? p.bias[n] : 0.0f;
      crow[n] = acc[nt][r] + bv;
    }
  }
}

// ---------------------------------------------------------------------------
// Fused attention reduction per (batch, head): S = scale*Q*K^T (WMMA) -> softmax
// row stats -> column weights w[m] -> sum_k (w.V)[k]*wo_rowsum[k] -> atomicAdd s[b]
// ---------------------------------------------------------------------------
__global__ __launch_bounds__(256) void attn_reduce(const float* Q, const float* Km, const float* V,
                                                   const float* wsum, float* sA, int Ttok){
  __shared__ _Float16 q16[128*32];
  __shared__ _Float16 k16[128*32];
  __shared__ _Float16 v16s[128*32];
  __shared__ _Float16 s16[128*130];
  __shared__ float mxs[128], invs[128], wv[128], oc[32];

  int tid = threadIdx.x;
  int bh = blockIdx.x;
  int b = bh / H_; int h = bh - b*H_;
  const float* qb = Q  + (size_t)b*Ttok*256 + h*32;
  const float* kb = Km + (size_t)b*Ttok*256 + h*32;
  const float* vb = V  + (size_t)b*Ttok*256 + h*32;
  int Tpad = (Ttok + 15) & ~15;
  int ntile = Tpad >> 4;

  for (int idx = tid; idx < Tpad*32; idx += 256){
    int t = idx >> 5, c = idx & 31;
    float qv=0.f, kv=0.f, vv=0.f;
    if (t < Ttok){ qv = qb[t*256+c]; kv = kb[t*256+c]; vv = vb[t*256+c]; }
    q16[idx]=(_Float16)qv; k16[idx]=(_Float16)kv; v16s[idx]=(_Float16)vv;
  }
  __syncthreads();
  int wave = tid >> 5, lane = tid & 31;
  const float scale = 0.17677669529663687f;    // 1/sqrt(32)
  if (wave < ntile){
    v16h a; load_frag_a(q16, 32, wave*16, 0, a);
    for (int nt = 0; nt < ntile; ++nt){
      v16h bf; load_frag_a(k16, 32, nt*16, 0, bf);
      v8f c = {0.f,0.f,0.f,0.f,0.f,0.f,0.f,0.f};
      c = __builtin_amdgcn_wmma_f32_16x16x32_f16(false, a, false, bf, (short)0, c, false, false);
      int colb = nt*16 + (lane & 15);
      int rb = wave*16 + ((lane < 16) ? 0 : 8);
      #pragma unroll
      for (int r = 0; r < 8; ++r) s16[(rb+r)*130 + colb] = (_Float16)(c[r]*scale);
    }
  }
  __syncthreads();
  for (int q = tid; q < Ttok; q += 256){
    float mx = -1e30f;
    for (int m = 0; m < Ttok; ++m) mx = fmaxf(mx, (float)s16[q*130+m]);
    float s = 0.f;
    for (int m = 0; m < Ttok; ++m) s += __expf((float)s16[q*130+m] - mx);
    mxs[q] = mx; invs[q] = 1.0f/s;
  }
  __syncthreads();
  for (int m = tid; m < Ttok; m += 256){
    float s = 0.f;
    for (int q = 0; q < Ttok; ++q) s += __expf((float)s16[q*130+m] - mxs[q]) * invs[q];
    wv[m] = s;
  }
  __syncthreads();
  if (tid < 32){
    float s = 0.f;
    for (int m = 0; m < Ttok; ++m) s += wv[m] * (float)v16s[m*32 + tid];
    oc[tid] = s * wsum[h*32 + tid];
  }
  __syncthreads();
  if (tid == 0){
    float s = 0.f;
    #pragma unroll
    for (int k = 0; k < 32; ++k) s += oc[k];
    atomicAdd(&sA[b], s);
  }
}

// wsum[(l*2+which)*257 + j]: j<256 -> sum_d wo[l][j][d]; j==256 -> sum_d bo[l][d]
__global__ __launch_bounds__(256) void wosum_kernel(const float* wo0, const float* bo0,
                                                    const float* wo1, const float* bo1, float* wsum){
  int bid = blockIdx.x;
  int l2 = bid / 257; int j = bid - l2*257;
  int l = l2 >> 1; int which = l2 & 1;
  const float* wo = which ? wo1 : wo0;
  const float* bo = which ? bo1 : bo0;
  const float* src = (j < 256) ? (wo + (size_t)l*65536 + (size_t)j*256) : (bo + (size_t)l*256);
  __shared__ float red[256];
  red[threadIdx.x] = src[threadIdx.x];
  __syncthreads();
  for (int s = 128; s > 0; s >>= 1){ if (threadIdx.x < s) red[threadIdx.x] += red[threadIdx.x+s]; __syncthreads(); }
  if (threadIdx.x == 0) wsum[bid] = red[0];
}

// s2[b] = sum relu((sA[b] + Ttok*bo_sum) * w0)
__global__ __launch_bounds__(256) void ff_kernel(const float* sA, const float* wsum257,
                                                 const float* w0, float Ttok, float* S2buf){
  int b = blockIdx.x;
  float s = sA[b] + Ttok * wsum257[256];
  float acc = 0.f;
  for (int e = threadIdx.x; e < T_*D_; e += 256) acc += fmaxf(s * w0[e], 0.0f);
  __shared__ float red[256];
  red[threadIdx.x] = acc; __syncthreads();
  for (int st = 128; st > 0; st >>= 1){ if (threadIdx.x < st) red[threadIdx.x] += red[threadIdx.x+st]; __syncthreads(); }
  if (threadIdx.x == 0) S2buf[b] = red[0];
}

// out[b,t,:] = LN(s2[b]*w1[t,:] + base[b,t,:]) * g + beta
__global__ __launch_bounds__(256) void ln_ff_kernel(const float* S2buf, const float* w1, const float* base,
                                                    const float* g, const float* beta, float* out){
  int bt = blockIdx.x; int b = bt >> 6; int t = bt & 63; int d = threadIdx.x;
  float val = S2buf[b] * w1[t*256 + d] + base[(size_t)bt*256 + d];
  __shared__ float r1[256], r2[256];
  r1[d]=val; r2[d]=val*val; __syncthreads();
  for (int s=128;s>0;s>>=1){ if (d<s){ r1[d]+=r1[d+s]; r2[d]+=r2[d+s]; } __syncthreads(); }
  float m = r1[0]*(1.0f/256.0f), var = r2[0]*(1.0f/256.0f) - m*m;
  out[(size_t)bt*256 + d] = (val - m)*rsqrtf(var + EPS_)*g[d] + beta[d];
}

__global__ __launch_bounds__(256) void ln_add_kernel(const float* A, const float* Bv,
                                                     const float* g, const float* beta, float* out){
  int bt = blockIdx.x; int d = threadIdx.x;
  float val = A[(size_t)bt*256+d] + Bv[(size_t)bt*256+d];
  __shared__ float r1[256], r2[256];
  r1[d]=val; r2[d]=val*val; __syncthreads();
  for (int s=128;s>0;s>>=1){ if (d<s){ r1[d]+=r1[d+s]; r2[d]+=r2[d+s]; } __syncthreads(); }
  float m = r1[0]*(1.0f/256.0f), var = r2[0]*(1.0f/256.0f) - m*m;
  out[(size_t)bt*256+d] = (val - m)*rsqrtf(var + EPS_)*g[d] + beta[d];
}

// ---------------------------------------------------------------------------
// GRU scan: 16 batches/block, 256 sequential steps. Per step h@Wh is a
// 16x192x64 GEMM: 6 waves x 2 N-tiles x 2 K-steps of WMMA; Wh fragments
// preloaded into registers once. Writes AP[b, dir*64+j, t] and final state.
// ---------------------------------------------------------------------------
__global__ __launch_bounds__(192) void gru_scan(const float* xmF, const float* xmB,
                                                const float* whF, const float* whB,
                                                const float* b1F, const float* b1B,
                                                const float* H0, float* AP, float* STATES, int layer){
  int gb0 = blockIdx.x * 16;
  int dir = blockIdx.y;
  const float* xm = dir ? xmB : xmF;
  const float* wh = dir ? whB : whF;
  const float* b1 = dir ? b1B : b1F;
  int tid = threadIdx.x, wave = tid >> 5, lane = tid & 31;

  __shared__ _Float16 hf[16*64];
  __shared__ float hv[16*64];
  __shared__ float mh[16*200];

  v16h bw[2][2];
  {
    int n_l = lane & 15;
    int kb = (lane < 16) ? 0 : 8;
    #pragma unroll
    for (int j = 0; j < 2; ++j){
      int n = (wave*2 + j)*16 + n_l;
      #pragma unroll
      for (int kt = 0; kt < 2; ++kt){
        #pragma unroll
        for (int v = 0; v < 8; ++v){
          int kk = kt*32 + ((v>>2)<<4) + kb + ((v&3)<<1);
          bw[kt][j][2*v]   = (_Float16)wh[kk*192 + n];
          bw[kt][j][2*v+1] = (_Float16)wh[(kk+1)*192 + n];
        }
      }
    }
  }
  for (int idx = tid; idx < 16*64; idx += 192){
    int bl = idx >> 6, j = idx & 63;
    float h = H0[(size_t)(gb0+bl)*512 + (layer*2 + dir)*64 + j];
    hv[idx] = h; hf[idx] = (_Float16)h;
  }
  __syncthreads();

  for (int t = 0; t < 256; ++t){
    v16h a0, a1;
    load_frag_a(hf, 64, 0, 0,  a0);
    load_frag_a(hf, 64, 0, 32, a1);
    #pragma unroll
    for (int j = 0; j < 2; ++j){
      v8f c = {0.f,0.f,0.f,0.f,0.f,0.f,0.f,0.f};
      c = __builtin_amdgcn_wmma_f32_16x16x32_f16(false, a0, false, bw[0][j], (short)0, c, false, false);
      c = __builtin_amdgcn_wmma_f32_16x16x32_f16(false, a1, false, bw[1][j], (short)0, c, false, false);
      int col = (wave*2 + j)*16 + (lane & 15);
      int rb = (lane < 16) ? 0 : 8;
      #pragma unroll
      for (int r = 0; r < 8; ++r) mh[(rb + r)*200 + col] = c[r];
    }
    __syncthreads();
    for (int idx = tid; idx < 16*64; idx += 192){
      int bl = idx >> 6, j = idx & 63;
      int b = gb0 + bl;
      const float* xrow = xm + ((size_t)b*256 + t)*192;
      float xz = xrow[j], xr = xrow[64+j], xh = xrow[128+j];
      float hz = mh[bl*200 + j]       + b1[j];
      float hr = mh[bl*200 + 64 + j]  + b1[64+j];
      float hh = mh[bl*200 + 128 + j] + b1[128+j];
      float z = sigmoidf_(xz + hz);
      float r = sigmoidf_(xr + hr);
      float cc = tanhf_(xh + r*hh);
      float hn = z*hv[idx] + (1.0f - z)*cc;
      hv[idx] = hn; hf[idx] = (_Float16)hn;
      AP[((size_t)b*128 + dir*64 + j)*256 + t] = hn;
    }
    __syncthreads();
  }
  for (int idx = tid; idx < 16*64; idx += 192){
    int bl = idx >> 6, j = idx & 63;
    STATES[(size_t)(gb0+bl)*512 + (layer*2+dir)*64 + j] = hv[idx];
  }
}

// ---------------------------------------------------------------------------
__global__ void split_x(const float* x, float* last){
  size_t i = (size_t)blockIdx.x*256 + threadIdx.x;
  if (i >= (size_t)B_*T_*D_) return;
  int d = (int)(i & 255); size_t bt = i >> 8;
  int t = (int)(bt & 63); size_t b = bt >> 6;
  last[i] = x[b*16896 + t*264 + d];
}
__global__ void init_h0(const float* x, const float* state, float* H0){
  int i = blockIdx.x*256 + threadIdx.x;
  if (i >= B_*512) return;
  int b = i >> 9, j = i & 511;
  H0[i] = state[i] + x[(size_t)b*16896 + (j>>3)*264 + 256 + (j&7)];
}
__global__ void assemble(const float* last, const float* states, float* out){
  size_t i = (size_t)blockIdx.x*256 + threadIdx.x;
  size_t n1 = (size_t)B_*16896;
  if (i < n1){
    size_t b = i / 16896; int rem = (int)(i % 16896);
    int t = rem / 264, d = rem % 264;
    out[i] = (d < 256) ? last[(b*64 + t)*256 + d] : states[b*512 + (size_t)(d-256)*64 + t];
  } else if (i < n1 + (size_t)B_*512){
    out[i] = states[i - n1];
  }
}

// ---------------------------------------------------------------------------
extern "C" void kernel_launch(void* const* d_in, const int* in_sizes, int n_in,
                              void* d_out, int out_size, void* d_ws, size_t ws_size,
                              hipStream_t stream){
  (void)in_sizes; (void)n_in; (void)out_size; (void)ws_size;
  const float* x       = (const float*)d_in[0];
  const float* state   = (const float*)d_in[1];
  const float* conv_w4 = (const float*)d_in[2];
  const float* conv_b4 = (const float*)d_in[3];
  const float* conv_w3 = (const float*)d_in[4];
  const float* conv_b3 = (const float*)d_in[5];
  const float* conv_w2 = (const float*)d_in[6];
  const float* conv_b2 = (const float*)d_in[7];
  const float* conv_w1 = (const float*)d_in[8];
  const float* conv_b1 = (const float*)d_in[9];
  const float* mha0_wq = (const float*)d_in[10];
  const float* mha0_bq = (const float*)d_in[11];
  const float* mha0_wk = (const float*)d_in[12];
  const float* mha0_bk = (const float*)d_in[13];
  const float* mha0_wv = (const float*)d_in[14];
  const float* mha0_bv = (const float*)d_in[15];
  const float* mha0_wo = (const float*)d_in[16];
  const float* mha0_bo = (const float*)d_in[17];
  const float* ff0_w0  = (const float*)d_in[18];
  const float* ff0_w1  = (const float*)d_in[19];
  const float* ln0_g   = (const float*)d_in[20];
  const float* ln0_b   = (const float*)d_in[21];
  const float* gruf_wx = (const float*)d_in[22];
  const float* gruf_wh = (const float*)d_in[23];
  const float* gruf_b  = (const float*)d_in[24];
  const float* grub_wx = (const float*)d_in[25];
  const float* grub_wh = (const float*)d_in[26];
  const float* grub_b  = (const float*)d_in[27];
  const float* mha1_wq = (const float*)d_in[28];
  const float* mha1_bq = (const float*)d_in[29];
  const float* mha1_wk = (const float*)d_in[30];
  const float* mha1_bk = (const float*)d_in[31];
  const float* mha1_wv = (const float*)d_in[32];
  const float* mha1_bv = (const float*)d_in[33];
  const float* mha1_wo = (const float*)d_in[34];
  const float* mha1_bo = (const float*)d_in[35];
  const float* ff1_w0  = (const float*)d_in[36];
  const float* ff1_w1  = (const float*)d_in[37];
  const float* ln1_g   = (const float*)d_in[38];
  const float* ln1_b   = (const float*)d_in[39];
  const float* lnb_g   = (const float*)d_in[40];
  const float* lnb_b   = (const float*)d_in[41];

  float* ws = (float*)d_ws;
  const size_t SZ_BT = (size_t)B_*T_*D_;            // 4,194,304
  float* LAST   = ws;
  float* D0b    = LAST + SZ_BT;
  float* BYP    = D0b + SZ_BT;                      // 4 buffers
  float* CC     = BYP + 4*SZ_BT;                    // B*100*256
  float* Qb     = CC + (size_t)B_*100*256;
  float* Kb     = Qb + (size_t)B_*128*256;
  float* Vb     = Kb + (size_t)B_*128*256;
  float* AP     = Vb + (size_t)B_*128*256;
  float* H0     = AP + (size_t)B_*128*256;
  float* STATES = H0 + B_*512;
  float* SA     = STATES + B_*512;
  float* S2v    = SA + 256;
  float* WSUM   = S2v + 256;
  float* XMF    = Qb;                               // alias over Q/K/V (disjoint lifetime)
  float* XMB    = Qb + (size_t)B_*256*192;

  auto gemm = [&](const float* A, const float* W, const float* bias, float* C,
                  int M, int N, int Kd, int rpb, int mode,
                  int a_off, int a_sB, int a_sR, int a_sC, int ks, int os,
                  int c_off, int c_sB, int c_sR){
    GemmP p; p.A=A; p.W=W; p.bias=bias; p.C=C; p.M=M; p.N=N; p.Kd=Kd; p.rpb=rpb;
    p.mode=mode; p.a_off=a_off; p.a_sB=a_sB; p.a_sR=a_sR; p.a_sC=a_sC;
    p.ks=ks; p.os=os; p.c_off=c_off; p.c_sB=c_sB; p.c_sR=c_sR;
    dim3 g(N/64, M/128);
    gemm_wmma<<<g, 256, 0, stream>>>(p);
  };

  split_x<<<(B_*T_*D_+255)/256, 256, 0, stream>>>(x, LAST);
  init_h0<<<(B_*512+255)/256, 256, 0, stream>>>(x, state, H0);
  wosum_kernel<<<L_*2*257, 256, 0, stream>>>(mha0_wo, mha0_bo, mha1_wo, mha1_bo, WSUM);

  int bi = 0;
  for (int i = 0; i < L_; ++i){
    // conv stack -> CC (B,100,256); row offsets 0/1/9/36
    gemm(LAST, conv_w4 + (size_t)i*64*65536, conv_b4 + i*256, CC, B_*1,  256, 64*256, 1,  2, 0,0,0,0, 4,1,  0*256, 100*256, 256);
    gemm(LAST, conv_w3 + (size_t)i*27*65536, conv_b3 + i*256, CC, B_*8,  256, 27*256, 8,  2, 0,0,0,0, 3,2,  1*256, 100*256, 256);
    gemm(LAST, conv_w2 + (size_t)i*8*65536,  conv_b2 + i*256, CC, B_*27, 256, 8*256,  27, 2, 0,0,0,0, 2,3,  9*256, 100*256, 256);
    gemm(LAST, conv_w1 + (size_t)i*65536,    conv_b1 + i*256, CC, B_*64, 256, 256,    64, 2, 0,0,0,0, 1,4, 36*256, 100*256, 256);
    // MHA0 QKV (Ttok=100)
    gemm(CC, mha0_wq + (size_t)i*65536, mha0_bq + i*256, Qb, B_*100, 256, 256, 100, 0, 0, 100*256, 256, 1, 0,0, 0, 100*256, 256);
    gemm(CC, mha0_wk + (size_t)i*65536, mha0_bk + i*256, Kb, B_*100, 256, 256, 100, 0, 0, 100*256, 256, 1, 0,0, 0, 100*256, 256);
    gemm(CC, mha0_wv + (size_t)i*65536, mha0_bv + i*256, Vb, B_*100, 256, 256, 100, 0, 0, 100*256, 256, 1, 0,0, 0, 100*256, 256);
    hipMemsetAsync(SA, 0, B_*sizeof(float), stream);
    attn_reduce<<<B_*H_, 256, 0, stream>>>(Qb, Kb, Vb, WSUM + (i*2+0)*257, SA, 100);
    ff_kernel<<<B_, 256, 0, stream>>>(SA, WSUM + (i*2+0)*257, ff0_w0 + (size_t)i*16384, 100.0f, S2v);
    ln_ff_kernel<<<B_*T_, 256, 0, stream>>>(S2v, ff0_w1 + (size_t)i*16384, LAST, ln0_g + i*256, ln0_b + i*256, D0b);
    // GRU x@Wx (A = d0 transposed; backward reads time-reversed rows)
    gemm(D0b, gruf_wx + (size_t)i*64*192, gruf_b + (size_t)i*384, XMF, B_*256, 192, 64, 256, 0,   0, 16384,  1, 256, 0,0, 0, 256*192, 192);
    gemm(D0b, grub_wx + (size_t)i*64*192, grub_b + (size_t)i*384, XMB, B_*256, 192, 64, 256, 0, 255, 16384, -1, 256, 0,0, 0, 256*192, 192);
    gru_scan<<<dim3(B_/16, 2), 192, 0, stream>>>(XMF, XMB,
        gruf_wh + (size_t)i*64*192, grub_wh + (size_t)i*64*192,
        gruf_b + (size_t)i*384 + 192, grub_b + (size_t)i*384 + 192,
        H0, AP, STATES, i);
    // MHA1 QKV (Ttok=128) from AP
    gemm(AP, mha1_wq + (size_t)i*65536, mha1_bq + i*256, Qb, B_*128, 256, 256, 128, 0, 0, 128*256, 256, 1, 0,0, 0, 128*256, 256);
    gemm(AP, mha1_wk + (size_t)i*65536, mha1_bk + i*256, Kb, B_*128, 256, 256, 128, 0, 0, 128*256, 256, 1, 0,0, 0, 128*256, 256);
    gemm(AP, mha1_wv + (size_t)i*65536, mha1_bv + i*256, Vb, B_*128, 256, 256, 128, 0, 0, 128*256, 256, 1, 0,0, 0, 128*256, 256);
    hipMemsetAsync(SA, 0, B_*sizeof(float), stream);
    attn_reduce<<<B_*H_, 256, 0, stream>>>(Qb, Kb, Vb, WSUM + (i*2+1)*257, SA, 128);
    ff_kernel<<<B_, 256, 0, stream>>>(SA, WSUM + (i*2+1)*257, ff1_w0 + (size_t)i*16384, 128.0f, S2v);
    float* d1 = BYP + (size_t)i*SZ_BT;
    ln_ff_kernel<<<B_*T_, 256, 0, stream>>>(S2v, ff1_w1 + (size_t)i*16384, D0b, ln1_g + i*256, ln1_b + i*256, d1);
    // bypass LN chain
    const float* lastSrc = d1;
    int j = 1;
    while ((i + 1) % j == 0){
      const float* bb = BYP + (size_t)(i - j + 1)*SZ_BT;
      ln_add_kernel<<<B_*T_, 256, 0, stream>>>(bb, lastSrc, lnb_g + bi*256, lnb_b + bi*256, LAST);
      lastSrc = LAST;
      bi++; j *= 2;
    }
  }
  size_t ntot = (size_t)B_*16896 + (size_t)B_*512;
  assemble<<<(unsigned)((ntot + 255)/256), 256, 0, stream>>>(LAST, STATES, (float*)d_out);
}
